// OldStyleConstructor_75617194213667
// MI455X (gfx1250) — matrix operand order repaired and checked by reading, not compiled, not run on hardware
//
#include <hip/hip_runtime.h>

// =====================================================================
// GraphConstructor pipeline for MI455X (gfx1250, wave32, WMMA).
//
// Heavy GEMMs run on v_wmma_f32_16x16x32_bf16 with DOUBLE-BUFFERED LDS
// tiles staged by global_load_async_to_lds_b128 (ASYNCcnt), so tile t+1
// streams into LDS while tile t is consumed by the matrix pipes.
// Weights are converted f32->bf16 WITH TRANSPOSE so every weight GEMM
// uses the fully-async transB path (no scalar LDS transpose stores).
//
// Assumed d_in flatten order (jax tree flatten: dict keys sorted):
//   0: token_ids (int32, 1024)
//   1: We (1025*128)  2: Wn (512*256)  3: Wt (128*8)
//   4: be (128)       5: bn (256)      6: bt (8)
//   7: emb (32000*512)
//   8+16*li+k, k in: 0:W1 1:W2 2:Wk 3:Wo 4:Wq 5:Wv 6:b1 7:b2
//                    8:bk 9:bo 10:bq 11:bv 12:c1 13:c2 14:g1 15:g2
// =====================================================================

typedef unsigned short u16;
typedef __attribute__((ext_vector_type(8)))  u16      u16x8;
typedef __attribute__((ext_vector_type(16))) __bf16   v16bf;
typedef __attribute__((ext_vector_type(8)))  float    v8f;
typedef __attribute__((ext_vector_type(4)))  unsigned v4u;
typedef __attribute__((ext_vector_type(8)))  int      v8i;
typedef __attribute__((ext_vector_type(4)))  int      v4i;

union U16V16 { u16x8 h[2]; v16bf v; };

#define S_LEN   1024
#define D_MODEL 512
#define D_FF    2048
#define D_NODE  256
#define D_EDGE  128
#define N_HEADS 8
#define D_HEAD  64
#define THRESH  0.0012f
#define NODE_SZ (S_LEN * D_NODE)

#define AS1 __attribute__((address_space(1)))
#define AS3 __attribute__((address_space(3)))
typedef AS1 v4i* gv4p;   // global-AS int32x4*  (param 1 of async builtin)
typedef AS3 v4i* lv4p;   // LDS-AS int32x4*     (param 2 of async builtin)

#if __has_builtin(__builtin_amdgcn_global_load_async_to_lds_b128) && \
    __has_builtin(__builtin_amdgcn_s_wait_asynccnt)
#define HAVE_ASYNC 1
#endif
#if __has_builtin(__builtin_amdgcn_tensor_load_to_lds) && \
    __has_builtin(__builtin_amdgcn_s_wait_tensorcnt)
#define HAVE_TDM 1
#endif

__device__ inline u16 f2bf(float f) {
  unsigned u = __float_as_uint(f);
  return (u16)((u + 0x7FFFu + ((u >> 16) & 1u)) >> 16);
}

// ------------------------------------------------------------------
// Generic WMMA GEMM: C[M,N] = A[M,K](bf16) * B(bf16) (+bias) (+relu)
// transB==0: B is [K,N] row-major (stride ldb)   (only P*V uses this)
// transB==1: B is [N,K] row-major (stride ldb)   (Q*K^T and all weights)
// Requires K % 32 == 0.
// Block tile 128x128, 8 waves arranged 2(M) x 4(N), wave tile 64x32.
// Double-buffered LDS; A (and transB-B) staged via async-to-LDS.
// NOTE: row guards are always true for launches here (M=1024; transB
// N multiple of 128), so per-wave async counts are uniform and the
// s_wait_asynccnt thresholds below are exact.
// ------------------------------------------------------------------
#define BLK_M 128
#define BLK_N 128
#define BLK_K 32
#define LDSTR 40   // u16 row stride in LDS: 80B, keeps 16B alignment

__global__ __launch_bounds__(256) void gemm_bf16_kernel(
    float* __restrict__ C, const u16* __restrict__ A, const u16* __restrict__ B,
    const float* __restrict__ bias, int M, int N, int K,
    int lda, int ldb, int ldc, int transB, int relu)
{
  __shared__ __align__(16) u16 As[2][BLK_M * LDSTR];
  __shared__ __align__(16) u16 Bs[2][BLK_N * LDSTR];

  const int tid  = threadIdx.x;
  const int lane = tid & 31;
  const int wave = tid >> 5;          // 0..7
  const int wm   = wave & 1;          // 2 waves along M
  const int wn   = wave >> 1;         // 4 waves along N
  const int half = lane >> 4;         // 0..1
  const int l16  = lane & 15;

  const int m0 = blockIdx.y * BLK_M;
  const int n0 = blockIdx.x * BLK_N;

  v8f acc[4][2];
  const v8f vzero = {0.f, 0.f, 0.f, 0.f, 0.f, 0.f, 0.f, 0.f};
#pragma unroll
  for (int i = 0; i < 4; ++i)
#pragma unroll
    for (int j = 0; j < 2; ++j) acc[i][j] = vzero;

  const u16x8 hzero = {0, 0, 0, 0, 0, 0, 0, 0};

  // staging roles
  const int ar  = tid >> 1;           // 0..127 (A row / transB-B row)
  const int asg = (tid & 1) * 16;     // 0 or 16 (k segment)
  const int bkk = tid >> 3;           // 0..31  (non-trans B: k row)
  const int bns = (tid & 7) * 16;     // 0..112 (non-trans B: n segment)

  u16x8 a0 = hzero, a1 = hzero, b0 = hzero, b1 = hzero;
  u16 bsc[16];

  // issue loads for tile starting at k0 into LDS buffer `buf`
  auto issue = [&](int buf, int k0) {
#if defined(HAVE_ASYNC)
    if (m0 + ar < M) {
      const u16* p = A + (size_t)(m0 + ar) * lda + k0 + asg;
      __builtin_amdgcn_global_load_async_to_lds_b128(
          (gv4p)(unsigned long long)(size_t)p,
          (lv4p)(unsigned)(size_t)&As[buf][ar * LDSTR + asg], 0, 0);
      __builtin_amdgcn_global_load_async_to_lds_b128(
          (gv4p)(unsigned long long)(size_t)(p + 8),
          (lv4p)(unsigned)(size_t)&As[buf][ar * LDSTR + asg + 8], 0, 0);
      __builtin_prefetch(p + BLK_K, 0, 3);
    }
    if (transB) {
      if (n0 + ar < N) {
        const u16* p = B + (size_t)(n0 + ar) * ldb + k0 + asg;
        __builtin_amdgcn_global_load_async_to_lds_b128(
            (gv4p)(unsigned long long)(size_t)p,
            (lv4p)(unsigned)(size_t)&Bs[buf][ar * LDSTR + asg], 0, 0);
        __builtin_amdgcn_global_load_async_to_lds_b128(
            (gv4p)(unsigned long long)(size_t)(p + 8),
            (lv4p)(unsigned)(size_t)&Bs[buf][ar * LDSTR + asg + 8], 0, 0);
      }
    } else {
      const u16* p = B + (size_t)(k0 + bkk) * ldb + n0 + bns;
#pragma unroll
      for (int j = 0; j < 16; ++j)
        bsc[j] = (n0 + bns + j < N) ? p[j] : (u16)0;
    }
#else
    a0 = hzero; a1 = hzero;
    if (m0 + ar < M) {
      const u16* p = A + (size_t)(m0 + ar) * lda + k0 + asg;
      a0 = *(const u16x8*)p;
      a1 = *(const u16x8*)(p + 8);
      __builtin_prefetch(p + BLK_K, 0, 3);
    }
    if (transB) {
      b0 = hzero; b1 = hzero;
      if (n0 + ar < N) {
        const u16* p = B + (size_t)(n0 + ar) * ldb + k0 + asg;
        b0 = *(const u16x8*)p;
        b1 = *(const u16x8*)(p + 8);
      }
    } else {
      const u16* p = B + (size_t)(k0 + bkk) * ldb + n0 + bns;
#pragma unroll
      for (int j = 0; j < 16; ++j)
        bsc[j] = (n0 + bns + j < N) ? p[j] : (u16)0;
    }
#endif
  };

  const int T = K / BLK_K;
  issue(0, 0);

  for (int t = 0; t < T; ++t) {
    const int cur = t & 1;
    // ---- commit register-staged data for tile t ----
#if defined(HAVE_ASYNC)
    if (!transB) {
#pragma unroll
      for (int j = 0; j < 16; ++j) Bs[cur][(bns + j) * LDSTR + bkk] = bsc[j];
    }
#else
    *(u16x8*)&As[cur][ar * LDSTR + asg]     = a0;
    *(u16x8*)&As[cur][ar * LDSTR + asg + 8] = a1;
    if (transB) {
      *(u16x8*)&Bs[cur][ar * LDSTR + asg]     = b0;
      *(u16x8*)&Bs[cur][ar * LDSTR + asg + 8] = b1;
    } else {
#pragma unroll
      for (int j = 0; j < 16; ++j) Bs[cur][(bns + j) * LDSTR + bkk] = bsc[j];
    }
#endif
    // ---- issue loads for tile t+1 into the other buffer ----
    if (t + 1 < T) issue(cur ^ 1, (t + 1) * BLK_K);
#if defined(HAVE_ASYNC)
    // tile t's async group must have landed; tile t+1's may stay in flight
    if (t + 1 < T) {
      if (transB) __builtin_amdgcn_s_wait_asynccnt(4);
      else        __builtin_amdgcn_s_wait_asynccnt(2);
    } else {
      __builtin_amdgcn_s_wait_asynccnt(0);
    }
#endif
    __syncthreads();

    // ---- WMMA: wave computes 4x2 tiles of 16x16 ----
    const u16* Ac = As[cur];
    const u16* Bc = Bs[cur];
    const int kbA = half * 8;   // A frag: lanes0-15 K{0..7,16..23}, lanes16-31 K{8..15,24..31}
    const int kbB = half * 8;   // B frag: lanes0-15 K{0..7,16..23}, lanes16-31 K{8..15,24..31}
    U16V16 af[4], bf[2];
#pragma unroll
    for (int j = 0; j < 2; ++j) {
      const int bn_ = wn * 32 + j * 16 + l16;
      bf[j].h[0] = *(const u16x8*)&Bc[bn_ * LDSTR + kbB];
      bf[j].h[1] = *(const u16x8*)&Bc[bn_ * LDSTR + kbB + 16];
    }
#pragma unroll
    for (int i = 0; i < 4; ++i) {
      const int am = wm * 64 + i * 16 + l16;
      af[i].h[0] = *(const u16x8*)&Ac[am * LDSTR + kbA];
      af[i].h[1] = *(const u16x8*)&Ac[am * LDSTR + kbA + 16];
#pragma unroll
      for (int j = 0; j < 2; ++j)
        acc[i][j] = __builtin_amdgcn_wmma_f32_16x16x32_bf16(
            false, af[i].v, false, bf[j].v, (short)0, acc[i][j], false, false);
    }
    __syncthreads();
  }

  // ---- store: VGPR r of C tile -> row (r + half*8), col l16 ----
#pragma unroll
  for (int j = 0; j < 2; ++j) {
    const int col = n0 + wn * 32 + j * 16 + l16;
    if (col >= N) continue;
    const float bv = bias ? bias[col] : 0.f;
#pragma unroll
    for (int i = 0; i < 4; ++i) {
      const int rbase = m0 + wm * 64 + i * 16 + half * 8;
#pragma unroll
      for (int r = 0; r < 8; ++r) {
        const int row = rbase + r;
        if (row < M) {
          float v = acc[i][j][r] + bv;
          if (relu) v = fmaxf(v, 0.f);
          C[(size_t)row * ldc + col] = v;
        }
      }
    }
  }
}

// ------------------------------------------------------------------
// CDNA5 TDM probe kernel (not launched; compile+histogram probe).
// 6-arg amdgpu-toolchain form: (v4u, v8i, v4i, v4i, v8i, i32 cpol).
// ------------------------------------------------------------------
__global__ __launch_bounds__(64) void cdna5_probe_kernel(const float* __restrict__ g,
                                                         float* __restrict__ out) {
  __shared__ __align__(16) float lbuf[1024];
#if defined(HAVE_TDM)
  if (threadIdx.x == 0) {
    // D# group0: count=1, lds_addr, global_addr, type=2 ("image")
    const unsigned lds_off = (unsigned)(size_t)&lbuf[0];  // low 32 bits = LDS offset
    const unsigned long long ga = (unsigned long long)(size_t)g;
    v4u g0;
    g0[0] = 1u;                                   // count=1, is_restore=0, gather=0
    g0[1] = lds_off;                              // lds_addr
    g0[2] = (unsigned)ga;                         // global_addr[31:0]
    g0[3] = (unsigned)((ga >> 32) & 0x1FFFFFFu) | (2u << 30);  // addr[56:32] | type=2
    // D# group1: data_size=4B, tensor 256x64 f32, tile 64x16, stride0=256
    v8i g1;
    g1[0] = (int)(2u << 16);                      // data_size=2 (4 bytes)
    g1[1] = (int)((256u & 0xFFFFu) << 16);        // tensor_dim0[15:0]
    g1[2] = (int)((256u >> 16) | ((64u & 0xFFFFu) << 16));  // dim0 hi | dim1 lo
    g1[3] = (int)((64u >> 16) | (64u << 16));     // dim1 hi | tile_dim0=64
    g1[4] = (int)16u;                             // tile_dim1=16, tile_dim2=0
    g1[5] = (int)256u;                            // tensor_dim0_stride lo
    g1[6] = 0;                                    // stride0 hi | stride1 lo
    g1[7] = 0;
    v4i g2 = {0, 0, 0, 0};
    v4i g3 = {0, 0, 0, 0};
    v8i g4 = {0, 0, 0, 0, 0, 0, 0, 0};
    __builtin_amdgcn_tensor_load_to_lds(g0, g1, g2, g3, g4, 0);
  }
  __builtin_amdgcn_s_wait_tensorcnt(0);
  __syncthreads();
#endif
#if defined(HAVE_ASYNC)
  {
    const float* gp = g + threadIdx.x * 4 + 4096;
    __builtin_amdgcn_global_load_async_to_lds_b128(
        (gv4p)(unsigned long long)(size_t)gp,
        (lv4p)(unsigned)(size_t)&lbuf[threadIdx.x * 4], 0, 0);
    __builtin_amdgcn_s_wait_asynccnt(0);
    __syncthreads();
  }
#endif
  float v = lbuf[threadIdx.x];
#if __has_builtin(__builtin_amdgcn_cluster_id_x)
  v += (float)__builtin_amdgcn_cluster_id_x();
#endif
  out[threadIdx.x] = v;
}

// ------------------------------------------------------------------
// Elementwise / fused kernels
// ------------------------------------------------------------------
__global__ void f32_to_bf16_kernel(const float* __restrict__ in, u16* __restrict__ out, int n) {
  int i = blockIdx.x * 256 + threadIdx.x;
  if (i < n) out[i] = f2bf(in[i]);
}

// WT[n*K + k] = bf16(W[k*N + n]) ; tiled 32x32 transpose, coalesced I/O
__global__ __launch_bounds__(256) void transpose_bf16_kernel(
    const float* __restrict__ in, u16* __restrict__ out, int K, int N) {
  __shared__ u16 tile[32][33];
  const int tx = threadIdx.x & 31, ty = threadIdx.x >> 5;   // ty 0..7
  const int k0 = blockIdx.y * 32, n0 = blockIdx.x * 32;
#pragma unroll
  for (int i = 0; i < 4; ++i) {
    const int k = k0 + ty + i * 8, n = n0 + tx;
    tile[ty + i * 8][tx] = (k < K && n < N) ? f2bf(in[(size_t)k * N + n]) : (u16)0;
  }
  __syncthreads();
#pragma unroll
  for (int i = 0; i < 4; ++i) {
    const int n = n0 + ty + i * 8, k = k0 + tx;
    if (n < N && k < K) out[(size_t)n * K + k] = tile[tx][ty + i * 8];
  }
}

__global__ void zero_f32_kernel(float* __restrict__ p, int n) {
  int i = blockIdx.x * 256 + threadIdx.x;
  if (i < n) p[i] = 0.f;
}

// x[s,d] = emb[tok[s],d] + PE(s,d); also bf16 copy
__global__ void embed_kernel(const int* __restrict__ tok, const float* __restrict__ emb,
                             float* __restrict__ x, u16* __restrict__ xb) {
  int idx = blockIdx.x * 256 + threadIdx.x;      // S*D
  int s = idx >> 9, d = idx & 511;
  float v = emb[(size_t)tok[s] * D_MODEL + d];
  int de = d & ~1;
  float ang = (float)s * __expf(-(float)de * (9.2103403719761836f / 512.0f));
  v += (d & 1) ? __cosf(ang) : __sinf(ang);
  x[idx] = v;
  xb[idx] = f2bf(v);
}

// row softmax of scores*0.125; writes bf16 P; accumulates P/8 into head-avg map
__global__ __launch_bounds__(256) void softmax_kernel(
    const float* __restrict__ scores, u16* __restrict__ Pb, float* __restrict__ acc) {
  __shared__ float red[256];
  const int r = blockIdx.x, tid = threadIdx.x;
  const float* row = scores + (size_t)r * S_LEN;
  float mx = -1e30f;
  for (int c = tid; c < S_LEN; c += 256) mx = fmaxf(mx, row[c] * 0.125f);
  red[tid] = mx; __syncthreads();
  for (int o = 128; o > 0; o >>= 1) { if (tid < o) red[tid] = fmaxf(red[tid], red[tid + o]); __syncthreads(); }
  mx = red[0]; __syncthreads();
  float sum = 0.f;
  for (int c = tid; c < S_LEN; c += 256) sum += __expf(row[c] * 0.125f - mx);
  red[tid] = sum; __syncthreads();
  for (int o = 128; o > 0; o >>= 1) { if (tid < o) red[tid] += red[tid + o]; __syncthreads(); }
  const float inv = 1.0f / red[0];
  for (int c = tid; c < S_LEN; c += 256) {
    float p = __expf(row[c] * 0.125f - mx) * inv;
    Pb[(size_t)r * S_LEN + c] = f2bf(p);
    acc[(size_t)r * S_LEN + c] += p * (1.0f / (float)N_HEADS);
  }
}

// out = LN(x + h) * g + b ; f32 + bf16 outputs. D_MODEL=512, 256 thr x 2 elems
__global__ __launch_bounds__(256) void ln_kernel(
    const float* __restrict__ x, const float* __restrict__ h,
    const float* __restrict__ g, const float* __restrict__ b,
    float* __restrict__ out, u16* __restrict__ outb) {
  __shared__ float red[256];
  const int r = blockIdx.x, tid = threadIdx.x;
  const size_t base = (size_t)r * D_MODEL;
  float y0 = x[base + tid] + h[base + tid];
  float y1 = x[base + tid + 256] + h[base + tid + 256];
  red[tid] = y0 + y1; __syncthreads();
  for (int o = 128; o > 0; o >>= 1) { if (tid < o) red[tid] += red[tid + o]; __syncthreads(); }
  const float m = red[0] * (1.0f / 512.0f); __syncthreads();
  const float d0 = y0 - m, d1 = y1 - m;
  red[tid] = d0 * d0 + d1 * d1; __syncthreads();
  for (int o = 128; o > 0; o >>= 1) { if (tid < o) red[tid] += red[tid + o]; __syncthreads(); }
  const float inv = rsqrtf(red[0] * (1.0f / 512.0f) + 1e-5f);
  const float o0 = d0 * inv * g[tid] + b[tid];
  const float o1 = d1 * inv * g[tid + 256] + b[tid + 256];
  out[base + tid] = o0;        out[base + tid + 256] = o1;
  outb[base + tid] = f2bf(o0); outb[base + tid + 256] = f2bf(o1);
}

__global__ void avg2_kernel(const float* __restrict__ a, const float* __restrict__ b,
                            float* __restrict__ out, u16* __restrict__ outb, int n) {
  int i = blockIdx.x * 256 + threadIdx.x;
  if (i < n) { float v = 0.5f * (a[i] + b[i]); out[i] = v; outb[i] = f2bf(v); }
}

// ------------------------------------------------------------------
// Edge extraction: mask (a0>T || a1>T, s!=t) -> count -> scan -> scatter
// row-major flat-id order matches jnp.unique's sorted output.
// ------------------------------------------------------------------
__device__ inline int edge_mask(const float* a0, const float* a1, int idx) {
  int s = idx >> 10, t = idx & 1023;
  return (s != t) && (a0[idx] > THRESH || a1[idx] > THRESH);
}

__global__ __launch_bounds__(256) void edge_count_kernel(
    const float* __restrict__ a0, const float* __restrict__ a1, unsigned* __restrict__ bcnt) {
  __shared__ unsigned red[256];
  const int idx = blockIdx.x * 256 + threadIdx.x;
  red[threadIdx.x] = (unsigned)edge_mask(a0, a1, idx);
  __syncthreads();
  for (int o = 128; o > 0; o >>= 1) { if (threadIdx.x < o) red[threadIdx.x] += red[threadIdx.x + o]; __syncthreads(); }
  if (threadIdx.x == 0) bcnt[blockIdx.x] = red[0];
}

// single block: exclusive scan over 4096 block counts; total -> Eout
__global__ __launch_bounds__(256) void scan_kernel(
    const unsigned* __restrict__ cnt, unsigned* __restrict__ offs, unsigned* __restrict__ Eout) {
  __shared__ unsigned part[256];
  const int tid = threadIdx.x;
  unsigned local[16], s = 0;
#pragma unroll
  for (int i = 0; i < 16; ++i) { local[i] = cnt[tid * 16 + i]; s += local[i]; }
  part[tid] = s; __syncthreads();
  for (int o = 1; o < 256; o <<= 1) {
    unsigned add = (tid >= o) ? part[tid - o] : 0u;
    __syncthreads();
    part[tid] += add;
    __syncthreads();
  }
  unsigned base = part[tid] - s;   // exclusive prefix
#pragma unroll
  for (int i = 0; i < 16; ++i) { offs[tid * 16 + i] = base; base += local[i]; }
  if (tid == 255) *Eout = part[255];
}

__global__ __launch_bounds__(256) void edge_scatter_kernel(
    const float* __restrict__ a0, const float* __restrict__ a1,
    const unsigned* __restrict__ offs, int* __restrict__ srcA, int* __restrict__ tgtA) {
  __shared__ unsigned sc[256];
  const int idx = blockIdx.x * 256 + threadIdx.x;
  const int tid = threadIdx.x;
  const unsigned m = (unsigned)edge_mask(a0, a1, idx);
  sc[tid] = m; __syncthreads();
  for (int o = 1; o < 256; o <<= 1) {
    unsigned add = (tid >= o) ? sc[tid - o] : 0u;
    __syncthreads();
    sc[tid] += add;
    __syncthreads();
  }
  if (m) {
    unsigned pos = offs[blockIdx.x] + sc[tid] - m;
    srcA[pos] = idx >> 10;
    tgtA[pos] = idx & 1023;
  }
}

// edge[e,:] = Pn[src] + Qn[tgt] + a*We_last + be   (written into d_out)
__global__ __launch_bounds__(256) void edge_feat_kernel(
    const int* __restrict__ srcA, const int* __restrict__ tgtA, const unsigned* __restrict__ Eptr,
    const float* __restrict__ Pn, const float* __restrict__ Qn,
    const float* __restrict__ a0, const float* __restrict__ a1,
    const float* __restrict__ WeLast, const float* __restrict__ be, float* __restrict__ out) {
  const unsigned E = *Eptr;
  const unsigned idx = blockIdx.x * 256 + threadIdx.x;
  const unsigned e = idx >> 7, j = idx & 127;
  if (e >= E) return;
  const int s = srcA[e], t = tgtA[e];
  const size_t aidx = (size_t)s * S_LEN + t;
  const float a = 0.5f * (a0[aidx] + a1[aidx]);
  out[NODE_SZ + (size_t)e * D_EDGE + j] =
      Pn[s * D_EDGE + j] + Qn[t * D_EDGE + j] + a * WeLast[j] + be[j];
}

__global__ void edge_index_kernel(const int* __restrict__ srcA, const int* __restrict__ tgtA,
                                  const unsigned* __restrict__ Eptr, float* __restrict__ out) {
  const unsigned E = *Eptr;
  const unsigned e = blockIdx.x * 256 + threadIdx.x;
  if (e < E) {
    const size_t base = NODE_SZ + (size_t)E * D_EDGE;
    out[base + e]     = (float)srcA[e];
    out[base + E + e] = (float)tgtA[e];
  }
}

__global__ __launch_bounds__(256) void logits_kernel(
    const unsigned* __restrict__ Eptr, const float* __restrict__ Wt,
    const float* __restrict__ bt, float* __restrict__ out) {
  const unsigned E = *Eptr;
  const unsigned idx = blockIdx.x * 256 + threadIdx.x;
  const unsigned e = idx >> 3, t = idx & 7;
  if (e >= E) return;
  const float* er = out + NODE_SZ + (size_t)e * D_EDGE;
  float s = bt[t];
#pragma unroll
  for (int j = 0; j < D_EDGE; ++j) s += er[j] * Wt[j * 8 + t];
  out[NODE_SZ + (size_t)E * 130 + (size_t)e * 8 + t] = s;   // 128 + 2 per edge before logits
}

// ------------------------------------------------------------------
// Host orchestration
// ------------------------------------------------------------------
extern "C" void kernel_launch(void* const* d_in, const int* in_sizes, int n_in,
                              void* d_out, int out_size, void* d_ws, size_t ws_size,
                              hipStream_t stream) {
  const int*   tok  = (const int*)d_in[0];
  const float* We   = (const float*)d_in[1];
  const float* Wn   = (const float*)d_in[2];
  const float* Wt   = (const float*)d_in[3];
  const float* be   = (const float*)d_in[4];
  const float* bn   = (const float*)d_in[5];
  const float* bt   = (const float*)d_in[6];
  const float* emb  = (const float*)d_in[7];
  auto LP = [&](int li, int j) -> const float* { return (const float*)d_in[8 + li * 16 + j]; };
  // 0:W1 1:W2 2:Wk 3:Wo 4:Wq 5:Wv 6:b1 7:b2 8:bk 9:bo 10:bq 11:bv 12:c1 13:c2 14:g1 15:g2

  const int SD = S_LEN * D_MODEL, SS = S_LEN * S_LEN, SF = S_LEN * D_FF;

  char* cur = (char*)d_ws;
  auto alloc = [&](size_t bytes) -> void* {
    void* p = cur; cur += (bytes + 255) & ~(size_t)255; return p;
  };
  float* x_f  = (float*)alloc(SD * 4);  u16* x_b  = (u16*)alloc(SD * 2);
  float* q_f  = (float*)alloc(SD * 4);  u16* q_b  = (u16*)alloc(SD * 2);
  float* k_f  = (float*)alloc(SD * 4);  u16* k_b  = (u16*)alloc(SD * 2);
  float* v_f  = (float*)alloc(SD * 4);  u16* v_b  = (u16*)alloc(SD * 2);
  float* sc_f = (float*)alloc((size_t)SS * 4);
  u16*   P_b  = (u16*)alloc((size_t)SS * 2);
  float* aL0  = (float*)alloc((size_t)SS * 4);
  float* aL1  = (float*)alloc((size_t)SS * 4);
  float* ao_f = (float*)alloc(SD * 4);  u16* ao_b = (u16*)alloc(SD * 2);
  float* o_f  = (float*)alloc(SD * 4);
  float* x1_f = (float*)alloc(SD * 4);  u16* x1_b = (u16*)alloc(SD * 2);
  float* hf_f = (float*)alloc((size_t)SF * 4);
  u16*   hf_b = (u16*)alloc((size_t)SF * 2);
  float* y_f  = (float*)alloc(SD * 4);
  float* out0_f = (float*)alloc(SD * 4); u16* out0_b = (u16*)alloc(SD * 2);
  float* out1_f = (float*)alloc(SD * 4); u16* out1_b = (u16*)alloc(SD * 2);
  float* avg_f  = (float*)alloc(SD * 4); u16* avg_b  = (u16*)alloc(SD * 2);
  u16*   wbuf   = (u16*)alloc((size_t)D_MODEL * D_FF * 2);   // biggest weight (transposed), reused
  float* Pn     = (float*)alloc(S_LEN * D_EDGE * 4);
  float* Qn     = (float*)alloc(S_LEN * D_EDGE * 4);
  int*   srcA   = (int*)alloc((size_t)SS * 4);
  int*   tgtA   = (int*)alloc((size_t)SS * 4);
  unsigned* bcnt = (unsigned*)alloc(4096 * 4);
  unsigned* boff = (unsigned*)alloc(4096 * 4);
  unsigned* Ecnt = (unsigned*)alloc(4);
  (void)ws_size; (void)n_in; (void)in_sizes; (void)out_size;

  auto conv = [&](const float* in, u16* out, int n) {
    f32_to_bf16_kernel<<<(n + 255) / 256, 256, 0, stream>>>(in, out, n);
  };
  // W[K x N] -> bf16 W^T[N x K] (so GEMMs can use the async transB path)
  auto convT = [&](const float* in, u16* out, int K, int N) {
    dim3 g(N / 32, K / 32);
    transpose_bf16_kernel<<<g, 256, 0, stream>>>(in, out, K, N);
  };
  auto gemm = [&](float* C, const u16* A, const u16* B, const float* bias,
                  int M, int N, int K, int lda, int ldb, int ldc, int transB, int relu) {
    dim3 g((N + BLK_N - 1) / BLK_N, (M + BLK_M - 1) / BLK_M);
    gemm_bf16_kernel<<<g, 256, 0, stream>>>(C, A, B, bias, M, N, K, lda, ldb, ldc, transB, relu);
  };

  embed_kernel<<<SD / 256, 256, 0, stream>>>(tok, emb, x_f, x_b);

  float* xin_f = x_f; u16* xin_b = x_b;
  float* attn_of_layer[2] = {aL0, aL1};
  float* lay_out_f[2] = {out0_f, out1_f};
  u16*   lay_out_b[2] = {out0_b, out1_b};

  for (int li = 0; li < 2; ++li) {
    float* aL = attn_of_layer[li];
    zero_f32_kernel<<<SS / 256, 256, 0, stream>>>(aL, SS);

    convT(LP(li, 4), wbuf, D_MODEL, D_MODEL);   // Wq^T
    gemm(q_f, xin_b, wbuf, LP(li, 10), S_LEN, D_MODEL, D_MODEL, D_MODEL, D_MODEL, D_MODEL, 1, 0);
    convT(LP(li, 2), wbuf, D_MODEL, D_MODEL);   // Wk^T
    gemm(k_f, xin_b, wbuf, LP(li, 8),  S_LEN, D_MODEL, D_MODEL, D_MODEL, D_MODEL, D_MODEL, 1, 0);
    convT(LP(li, 5), wbuf, D_MODEL, D_MODEL);   // Wv^T
    gemm(v_f, xin_b, wbuf, LP(li, 11), S_LEN, D_MODEL, D_MODEL, D_MODEL, D_MODEL, D_MODEL, 1, 0);
    conv(q_f, q_b, SD); conv(k_f, k_b, SD); conv(v_f, v_b, SD);

    for (int h = 0; h < N_HEADS; ++h) {
      // scores = Q_h * K_h^T   (B = K rows, transB -> LDS [n][k] directly)
      gemm(sc_f, q_b + h * D_HEAD, k_b + h * D_HEAD, nullptr,
           S_LEN, S_LEN, D_HEAD, D_MODEL, D_MODEL, S_LEN, 1, 0);
      softmax_kernel<<<S_LEN, 256, 0, stream>>>(sc_f, P_b, aL);
      // O_h = P * V_h  (only non-trans user; A=P dominates traffic)
      gemm(ao_f + h * D_HEAD, P_b, v_b + h * D_HEAD, nullptr,
           S_LEN, D_HEAD, S_LEN, S_LEN, D_MODEL, D_MODEL, 0, 0);
    }
    conv(ao_f, ao_b, SD);
    convT(LP(li, 3), wbuf, D_MODEL, D_MODEL);   // Wo^T
    gemm(o_f, ao_b, wbuf, LP(li, 9), S_LEN, D_MODEL, D_MODEL, D_MODEL, D_MODEL, D_MODEL, 1, 0);
    ln_kernel<<<S_LEN, 256, 0, stream>>>(xin_f, o_f, LP(li, 14), LP(li, 6), x1_f, x1_b);

    convT(LP(li, 0), wbuf, D_MODEL, D_FF);      // W1^T [2048 x 512]
    gemm(hf_f, x1_b, wbuf, LP(li, 12), S_LEN, D_FF, D_MODEL, D_MODEL, D_MODEL, D_FF, 1, 1);
    conv(hf_f, hf_b, SF);
    convT(LP(li, 1), wbuf, D_FF, D_MODEL);      // W2^T [512 x 2048]
    gemm(y_f, hf_b, wbuf, LP(li, 13), S_LEN, D_MODEL, D_FF, D_FF, D_FF, D_MODEL, 1, 0);
    ln_kernel<<<S_LEN, 256, 0, stream>>>(x1_f, y_f, LP(li, 15), LP(li, 7), lay_out_f[li], lay_out_b[li]);

    xin_f = lay_out_f[li]; xin_b = lay_out_b[li];
  }

  // avg_emb, node head, edge-head partials
  avg2_kernel<<<SD / 256, 256, 0, stream>>>(out0_f, out1_f, avg_f, avg_b, SD);
  convT(Wn, wbuf, D_MODEL, D_NODE);             // Wn^T [256 x 512]
  gemm((float*)d_out, out1_b, wbuf, bn, S_LEN, D_NODE, D_MODEL, D_MODEL, D_MODEL, D_NODE, 1, 0);
  // We rows 0..511 and 512..1023, each [512 x 128] -> transposed [128 x 512]
  convT(We,                           wbuf,                   D_MODEL, D_EDGE);
  convT(We + (size_t)D_MODEL * D_EDGE, wbuf + D_EDGE * D_MODEL, D_MODEL, D_EDGE);
  gemm(Pn, avg_b, wbuf,                     nullptr, S_LEN, D_EDGE, D_MODEL, D_MODEL, D_MODEL, D_EDGE, 1, 0);
  gemm(Qn, avg_b, wbuf + D_EDGE * D_MODEL,  nullptr, S_LEN, D_EDGE, D_MODEL, D_MODEL, D_MODEL, D_EDGE, 1, 0);

  // edges: threshold union -> sorted compaction
  edge_count_kernel<<<SS / 256, 256, 0, stream>>>(aL0, aL1, bcnt);
  scan_kernel<<<1, 256, 0, stream>>>(bcnt, boff, Ecnt);
  edge_scatter_kernel<<<SS / 256, 256, 0, stream>>>(aL0, aL1, boff, srcA, tgtA);

  const float* WeLast = We + (size_t)1024 * D_EDGE;
  edge_feat_kernel<<<(size_t)SS * D_EDGE / 256, 256, 0, stream>>>(
      srcA, tgtA, Ecnt, Pn, Qn, aL0, aL1, WeLast, be, (float*)d_out);
  edge_index_kernel<<<SS / 256, 256, 0, stream>>>(srcA, tgtA, Ecnt, (float*)d_out);
  logits_kernel<<<(size_t)SS * 8 / 256, 256, 0, stream>>>(Ecnt, Wt, bt, (float*)d_out);
}
